// MORN_SurvPathStyle_54709293416902
// MI455X (gfx1250) — matrix operand order repaired and checked by reading, not compiled, not run on hardware
//
#include <hip/hip_runtime.h>
#include <math.h>

// ---------------- sizes from the reference ----------------
#define HID   256
#define NBINS 4
#define NMIR  2000
#define NMRNA 20000
#define BATCH 16
#define NP    2048
#define PD    1024
#define EREG  300000
#define EPM   320000
#define EPI   32000

typedef __attribute__((ext_vector_type(16))) __bf16 v16bf;
typedef __attribute__((ext_vector_type(8)))  float  v8f;
typedef __attribute__((ext_vector_type(4)))  unsigned v4u;
typedef __attribute__((ext_vector_type(8)))  unsigned v8u;

// ---------------- bf16 helpers: native HW convert (RNE) ----------------
static __device__ __forceinline__ __bf16 f2bf(float f) { return (__bf16)f; }
static __device__ __forceinline__ float  bf2f(__bf16 h) { return (float)h; }
static __device__ __forceinline__ float gelu_exact(float x) {
  return 0.5f * x * (1.0f + erff(x * 0.70710678118654752f));
}

// ---------------- zero scratch ----------------
__global__ void zero_f32(float* __restrict__ p, long n) {
  long i = (long)blockIdx.x * blockDim.x + threadIdx.x;
  long stride = (long)gridDim.x * blockDim.x;
  for (; i < n; i += stride) p[i] = 0.0f;
}

// ---------------- split f32 weights into bf16 hi/lo ----------------
__global__ void split_w_bf16(const float* __restrict__ W,
                             __bf16* __restrict__ hi, __bf16* __restrict__ lo, int n) {
  int i = blockIdx.x * blockDim.x + threadIdx.x;
  if (i < n) {
    float x = W[i];
    __bf16 h = f2bf(x);
    hi[i] = h;
    lo[i] = f2bf(x - bf2f(h));
  }
}

// ---------------- COO weighted scatter:  out[dst] += h[src]*w  ----------------
#define EPB 8
__global__ __launch_bounds__(HID) void scatter_coo(
    const float* __restrict__ h, const float* __restrict__ w,
    const int* __restrict__ src, const int* __restrict__ dst,
    int nE, float* __restrict__ out) {
  int t = threadIdx.x;
  int e0 = blockIdx.x * EPB;
  int e1 = e0 + EPB; if (e1 > nE) e1 = nE;
  for (int e = e0; e < e1; ++e) {
    int s = src[e], d = dst[e];
    float val = h[(size_t)s * HID + t] * w[e];
    atomicAdd(&out[(size_t)d * HID + t], val);
  }
}

// ---------------- patient readout scatter, dst < 16: LDS binning ----------------
#define PAT_EPB 512
__global__ __launch_bounds__(HID) void scatter_pat(
    const float* __restrict__ h, const float* __restrict__ w,
    const int* __restrict__ src, const int* __restrict__ dst,
    int nE, float* __restrict__ acc) {
  __shared__ float lbin[BATCH * HID];
  int t = threadIdx.x;
  for (int i = t; i < BATCH * HID; i += HID) lbin[i] = 0.0f;
  __syncthreads();
  int e0 = blockIdx.x * PAT_EPB;
  int e1 = e0 + PAT_EPB; if (e1 > nE) e1 = nE;
  for (int e = e0; e < e1; ++e) {
    int s = src[e], d = dst[e];
    lbin[d * HID + t] += h[(size_t)s * HID + t] * w[e];
  }
  __syncthreads();
  for (int i = t; i < BATCH * HID; i += HID) atomicAdd(&acc[i], lbin[i]);
}

// ---------------- WMMA GEMM + GELU + residual (bf16x3 split for ~fp32 accuracy)
// out[m,n] = resid[m,n] + gelu( sum_k A[m,k]*W[n,k] + bias[n] ),  K = N = 256.
// A tile (16x256 f32, padded to pitch 260 floats) is staged into LDS by the
// Tensor Data Mover: 2D D# with pad_enable, pad_interval=7 (256 DWORDs),
// pad_amount=3 (4 DWORDs) -> exactly the 260-float conflict-free pitch.
__global__ __launch_bounds__(256) void wmma_update(
    const float*  __restrict__ A,     // [M,256]
    const __bf16* __restrict__ Whi,   // [256,256] row-major (n-major)
    const __bf16* __restrict__ Wlo,
    const float*  __restrict__ bias,  // [256]
    const float*  __restrict__ resid, // [M,256]
    float*        __restrict__ out,   // [M,256]
    int M) {
  __shared__ float sA[16 * 260];
  const int m0 = blockIdx.x * 16;
  const int t = threadIdx.x;
  const int wave = t >> 5;
  const int lane = t & 31;

  if (wave == 0) {
    // ---- TDM descriptor (ISA 8.3/8.4), issued once per block by wave 0 ----
    unsigned long long ga = (unsigned long long)(uintptr_t)(A + (size_t)m0 * HID);
    unsigned lds_off = (unsigned)(uintptr_t)(&sA[0]);   // LDS aperture: low 32 bits
    v4u g0;
    g0[0] = 1u;                                   // count=1, user mode
    g0[1] = lds_off;                              // lds_addr (bytes)
    g0[2] = (unsigned)ga;                         // global_addr[31:0]
    g0[3] = (unsigned)((ga >> 32) & 0x01FFFFFFu)  // global_addr[56:32]
            | (2u << 30);                         // type=2 (image)
    v8u g1;
    g1[0] = (2u << 16)      // data_size = 4B
          | (1u << 20)      // pad_enable
          | (7u << 22)      // pad_interval: 256 DWORDs
          | (3u << 25);     // pad_amount: 4 DWORDs
    g1[1] = (256u << 16);   // tensor_dim0 = 256 (low16 at bits 63:48)
    g1[2] = (16u  << 16);   // tensor_dim1 = 16  (low16 at bits 95:80)
    g1[3] = (256u << 16);   // tile_dim0 = 256   (bits 127:112)
    g1[4] = 16u;            // tile_dim1 = 16; tile_dim2 = 0
    g1[5] = 256u;           // tensor_dim0_stride = 256 (low 32)
    g1[6] = 0u;             // stride hi / tensor_dim1_stride lo
    g1[7] = 0u;
    asm volatile("tensor_load_to_lds %0, %1" : : "s"(g0), "s"(g1) : "memory");
    asm volatile("s_wait_tensorcnt 0x0" : : : "memory");
  }
  __syncthreads();

  const int half = lane >> 4;   // K half selector (A & B fragments)
  const int l16  = lane & 15;   // A: row m; B/C/D: column n

  v8f c0 = {}; v8f c1 = {};
  for (int kt = 0; kt < HID / 32; ++kt) {
    // --- A fragment (16x32 bf16, contiguous k = 16*half + s), split into hi/lo ---
    v16bf ahi, alo;
    const float* arow = &sA[l16 * 260 + kt * 32 + half * 16];
#pragma unroll
    for (int j = 0; j < 8; ++j) {
      float x = arow[j];
      __bf16 h = f2bf(x);
      ahi[j] = h; alo[j] = f2bf(x - bf2f(h));
      float y = arow[8 + j];
      __bf16 h2 = f2bf(y);
      ahi[8 + j] = h2; alo[8 + j] = f2bf(y - bf2f(h2));
    }
#pragma unroll
    for (int ti = 0; ti < 2; ++ti) {
      int nt = wave + ti * 8;
      int n  = nt * 16 + l16;
      const v16bf bhi = *(const v16bf*)(Whi + (size_t)n * HID + kt * 32 + half * 16);
      const v16bf blo = *(const v16bf*)(Wlo + (size_t)n * HID + kt * 32 + half * 16);
      v8f& c = ti ? c1 : c0;
      c = __builtin_amdgcn_wmma_f32_16x16x32_bf16(false, ahi, false, bhi, (short)0, c, false, false);
      c = __builtin_amdgcn_wmma_f32_16x16x32_bf16(false, ahi, false, blo, (short)0, c, false, false);
      c = __builtin_amdgcn_wmma_f32_16x16x32_bf16(false, alo, false, bhi, (short)0, c, false, false);
    }
  }
  // epilogue: C/D layout -> m = r + 8*half, n = nt*16 + l16
#pragma unroll
  for (int ti = 0; ti < 2; ++ti) {
    v8f c = ti ? c1 : c0;
    int n = (wave + ti * 8) * 16 + l16;
    float b = bias[n];
#pragma unroll
    for (int r = 0; r < 8; ++r) {
      int m = m0 + r + half * 8;
      float x = c[r] + b;
      out[(size_t)m * HID + n] = resid[(size_t)m * HID + n] + gelu_exact(x);
    }
  }
}

// ---------------- masked patch pooling: pooled_raw[b,d] += sum_p mask*patches ----
__global__ __launch_bounds__(HID) void pool_patches(
    const float* __restrict__ patches, const float* __restrict__ mask,
    float* __restrict__ pooled_raw) {
  __shared__ float sm[128];
  int bid = blockIdx.x;
  int ps  = bid & 15;
  int dch = (bid >> 4) & 3;
  int b   = bid >> 6;
  int t = threadIdx.x;
  if (t < 128) sm[t] = mask[b * NP + ps * 128 + t];
  __syncthreads();
  int d = dch * HID + t;
  float s = 0.0f;
  const float* base = patches + ((size_t)b * NP + (size_t)ps * 128) * PD + d;
  for (int p = 0; p < 128; ++p) s += base[(size_t)p * PD] * sm[p];
  atomicAdd(&pooled_raw[b * PD + d], s);
}

// ---------------- tail: all [16,*] heads, GELU, LayerNorm, fusion ----------------
__global__ __launch_bounds__(256) void tail_kernel(
    const float* __restrict__ acc,        // [16,256]
    const float* __restrict__ pooled_raw, // [16,1024] = sum_p mask*patches
    const float* __restrict__ mask,       // [16,2048]
    const float* __restrict__ proj_w, const float* __restrict__ proj_b,
    const float* __restrict__ ln_g,   const float* __restrict__ ln_b,
    const float* __restrict__ omics_w,const float* __restrict__ omics_b,
    const float* __restrict__ wsi_w,  const float* __restrict__ wsi_b,
    const float* __restrict__ mlp_w,  const float* __restrict__ mlp_b,
    const float* __restrict__ wln_g,  const float* __restrict__ wln_b,
    const float* __restrict__ fuse_w1,const float* __restrict__ fuse_b1,
    const float* __restrict__ fuse_w2,const float* __restrict__ fuse_b2,
    float* __restrict__ out) {
  __shared__ float s_pat[BATCH * HID];
  __shared__ float s_wsi[BATCH * HID];
  __shared__ float s_tmp[BATCH * HID];
  __shared__ float s_ms[BATCH], s_mu[BATCH], s_var[BATCH];
  int t = threadIdx.x;

  if (t < BATCH) {
    float s = 0.0f;
    for (int p = 0; p < NP; ++p) s += mask[t * NP + p];
    s_ms[t] = s;
  }
  __syncthreads();

  // WSI: pooled = (sum mask*patches)@wsi_w^T/denom + wsi_b*(masksum/denom); gelu -> s_tmp
  for (int i = t; i < BATCH * HID; i += 256) {
    int b = i >> 8, h = i & 255;
    float ms = s_ms[b];
    float den = ms < 1.0f ? 1.0f : ms;
    float s = 0.0f;
    const float* pr = &pooled_raw[b * PD];
    const float* w  = &wsi_w[(size_t)h * PD];
    for (int k = 0; k < PD; ++k) s += pr[k] * w[k];
    s_tmp[i] = gelu_exact(s / den + wsi_b[h] * (ms / den));
  }
  __syncthreads();
  // mlp + gelu -> s_wsi
  for (int i = t; i < BATCH * HID; i += 256) {
    int b = i >> 8, h = i & 255;
    float s = mlp_b[h];
    const float* g = &s_tmp[b * HID];
    const float* w = &mlp_w[h * HID];
    for (int k = 0; k < HID; ++k) s += g[k] * w[k];
    s_wsi[i] = gelu_exact(s);
  }
  __syncthreads();
  // LN(s_wsi)
  if (t < BATCH) {
    float m = 0.0f;
    for (int k = 0; k < HID; ++k) m += s_wsi[t * HID + k];
    m *= (1.0f / HID);
    float v = 0.0f;
    for (int k = 0; k < HID; ++k) { float d = s_wsi[t * HID + k] - m; v += d * d; }
    s_mu[t] = m; s_var[t] = v * (1.0f / HID);
  }
  __syncthreads();
  for (int i = t; i < BATCH * HID; i += 256) {
    int b = i >> 8, h = i & 255;
    s_wsi[i] = (s_wsi[i] - s_mu[b]) * rsqrtf(s_var[b] + 1e-5f) * wln_g[h] + wln_b[h];
  }
  __syncthreads();
  // pat: gelu(acc@proj_w^T + proj_b) -> s_pat
  for (int i = t; i < BATCH * HID; i += 256) {
    int b = i >> 8, h = i & 255;
    float s = proj_b[h];
    const float* a = &acc[b * HID];
    const float* w = &proj_w[h * HID];
    for (int k = 0; k < HID; ++k) s += a[k] * w[k];
    s_pat[i] = gelu_exact(s);
  }
  __syncthreads();
  // LN(s_pat)
  if (t < BATCH) {
    float m = 0.0f;
    for (int k = 0; k < HID; ++k) m += s_pat[t * HID + k];
    m *= (1.0f / HID);
    float v = 0.0f;
    for (int k = 0; k < HID; ++k) { float d = s_pat[t * HID + k] - m; v += d * d; }
    s_mu[t] = m; s_var[t] = v * (1.0f / HID);
  }
  __syncthreads();
  for (int i = t; i < BATCH * HID; i += 256) {
    int b = i >> 8, h = i & 255;
    s_pat[i] = (s_pat[i] - s_mu[b]) * rsqrtf(s_var[b] + 1e-5f) * ln_g[h] + ln_b[h];
  }
  __syncthreads();
  // fuse1 on cat=[pat, wsi] -> gelu -> s_tmp
  for (int i = t; i < BATCH * HID; i += 256) {
    int b = i >> 8, h = i & 255;
    float s = fuse_b1[h];
    const float* w = &fuse_w1[(size_t)h * 2 * HID];
    for (int k = 0; k < HID; ++k) s += s_pat[b * HID + k] * w[k];
    for (int k = 0; k < HID; ++k) s += s_wsi[b * HID + k] * w[HID + k];
    s_tmp[i] = gelu_exact(s);
  }
  __syncthreads();
  // outputs: fused [16,4] then omics [16,4]
  if (t < BATCH * NBINS) {
    int b = t >> 2, j = t & 3;
    float s = fuse_b2[j];
    const float* w = &fuse_w2[j * HID];
    for (int k = 0; k < HID; ++k) s += s_tmp[b * HID + k] * w[k];
    out[b * NBINS + j] = s;
    float s2 = omics_b[j];
    const float* w2 = &omics_w[j * HID];
    for (int k = 0; k < HID; ++k) s2 += s_pat[b * HID + k] * w2[k];
    out[BATCH * NBINS + b * NBINS + j] = s2;
  }
}

// ---------------- host launcher ----------------
extern "C" void kernel_launch(void* const* d_in, const int* in_sizes, int n_in,
                              void* d_out, int out_size, void* d_ws, size_t ws_size,
                              hipStream_t stream) {
  const float* emb_mirna  = (const float*)d_in[0];
  const float* emb_mrna   = (const float*)d_in[1];
  const float* w_mir2mrna = (const float*)d_in[2];
  const float* w_mrna2mir = (const float*)d_in[3];
  const float* w_mrna2pat = (const float*)d_in[4];
  const float* w_mir2pat  = (const float*)d_in[5];
  const float* reg_w_mrna = (const float*)d_in[6];
  const float* reg_b_mrna = (const float*)d_in[7];
  const float* reg_w_mir  = (const float*)d_in[8];
  const float* reg_b_mir  = (const float*)d_in[9];
  const float* proj_w = (const float*)d_in[10];
  const float* proj_b = (const float*)d_in[11];
  const float* ln_g   = (const float*)d_in[12];
  const float* ln_b   = (const float*)d_in[13];
  const float* omics_w= (const float*)d_in[14];
  const float* omics_b= (const float*)d_in[15];
  const float* wsi_w  = (const float*)d_in[16];
  const float* wsi_b  = (const float*)d_in[17];
  const float* mlp_w  = (const float*)d_in[18];
  const float* mlp_b  = (const float*)d_in[19];
  const float* wln_g  = (const float*)d_in[20];
  const float* wln_b  = (const float*)d_in[21];
  const float* fuse_w1= (const float*)d_in[22];
  const float* fuse_b1= (const float*)d_in[23];
  const float* fuse_w2= (const float*)d_in[24];
  const float* fuse_b2= (const float*)d_in[25];
  const float* patches= (const float*)d_in[26];
  const float* mask   = (const float*)d_in[27];
  const int* src_mir2mrna = (const int*)d_in[28];
  const int* dst_mir2mrna = (const int*)d_in[29];
  const int* src_mrna2mir = (const int*)d_in[30];
  const int* dst_mrna2mir = (const int*)d_in[31];
  const int* src_mrna2pat = (const int*)d_in[32];
  const int* dst_mrna2pat = (const int*)d_in[33];
  const int* src_mir2pat  = (const int*)d_in[34];
  const int* dst_mir2pat  = (const int*)d_in[35];

  // ---- workspace layout (all offsets 256B aligned) ----
  char* ws = (char*)d_ws;
  size_t off = 0;
  auto take = [&](size_t bytes) -> char* {
    char* p = ws + off;
    off = (off + bytes + 255) & ~(size_t)255;
    return p;
  };
  float* agg_mrna   = (float*)take((size_t)NMRNA * HID * 4);
  float* agg_mir    = (float*)take((size_t)NMIR  * HID * 4);
  float* acc        = (float*)take((size_t)BATCH * HID * 4);
  float* pooled_raw = (float*)take((size_t)BATCH * PD  * 4);
  long   nZero = (long)((float*)(ws + off) - agg_mrna);
  float* h_mrna2 = (float*)take((size_t)NMRNA * HID * 4);
  float* h_mir2  = (float*)take((size_t)NMIR  * HID * 4);
  __bf16* whi_mrna = (__bf16*)take((size_t)HID * HID * 2);
  __bf16* wlo_mrna = (__bf16*)take((size_t)HID * HID * 2);
  __bf16* whi_mir  = (__bf16*)take((size_t)HID * HID * 2);
  __bf16* wlo_mir  = (__bf16*)take((size_t)HID * HID * 2);

  // 1) zero accumulators
  zero_f32<<<4096, 256, 0, stream>>>(agg_mrna, nZero);

  // 2) split update-GEMM weights into bf16 hi/lo
  split_w_bf16<<<(HID * HID + 255) / 256, 256, 0, stream>>>(reg_w_mrna, whi_mrna, wlo_mrna, HID * HID);
  split_w_bf16<<<(HID * HID + 255) / 256, 256, 0, stream>>>(reg_w_mir,  whi_mir,  wlo_mir,  HID * HID);

  // 3) regulatory scatters on ORIGINAL embeddings
  scatter_coo<<<(EREG + EPB - 1) / EPB, HID, 0, stream>>>(emb_mirna, w_mir2mrna, src_mir2mrna, dst_mir2mrna, EREG, agg_mrna);
  scatter_coo<<<(EREG + EPB - 1) / EPB, HID, 0, stream>>>(emb_mrna,  w_mrna2mir, src_mrna2mir, dst_mrna2mir, EREG, agg_mir);

  // 4) WMMA node updates: h' = emb + gelu(agg @ W^T + b)  (A tile via TDM)
  wmma_update<<<NMRNA / 16, 256, 0, stream>>>(agg_mrna, whi_mrna, wlo_mrna, reg_b_mrna, emb_mrna, h_mrna2, NMRNA);
  wmma_update<<<NMIR  / 16, 256, 0, stream>>>(agg_mir,  whi_mir,  wlo_mir,  reg_b_mir,  emb_mirna, h_mir2, NMIR);

  // 5) patient readout (LDS-binned) on updated embeddings
  scatter_pat<<<(EPM + PAT_EPB - 1) / PAT_EPB, HID, 0, stream>>>(h_mrna2, w_mrna2pat, src_mrna2pat, dst_mrna2pat, EPM, acc);
  scatter_pat<<<(EPI + PAT_EPB - 1) / PAT_EPB, HID, 0, stream>>>(h_mir2,  w_mir2pat,  src_mir2pat,  dst_mir2pat,  EPI, acc);

  // 6) masked patch pooling (linear commutes with masked mean)
  pool_patches<<<BATCH * 4 * 16, HID, 0, stream>>>(patches, mask, pooled_raw);

  // 7) all small heads + fusion
  tail_kernel<<<1, 256, 0, stream>>>(acc, pooled_raw, mask,
                                     proj_w, proj_b, ln_g, ln_b, omics_w, omics_b,
                                     wsi_w, wsi_b, mlp_w, mlp_b, wln_g, wln_b,
                                     fuse_w1, fuse_b1, fuse_w2, fuse_b2,
                                     (float*)d_out);
}
